// VqGanCLIPGenerator_41180146434535
// MI455X (gfx1250) — matrix operand order repaired and checked by reading, not compiled, not run on hardware
//
#include <hip/hip_runtime.h>
#include <hip/hip_bf16.h>

// ---------------- types for WMMA ----------------
typedef __attribute__((ext_vector_type(16))) __bf16 v16bf;
typedef __attribute__((ext_vector_type(8)))  float  v8f;

#define M_TOTAL 8192        // B*H*W = 8*32*32
#define NE      16384       // codebook entries
#define E       256         // embedding dim
#define NSEG    4           // N-dimension split for occupancy
#define SEG_N   (NE / NSEG)           // 4096
#define TILES_PER_SEG (SEG_N / 16)    // 256
#define LDS_PAD 8                     // 8 bf16 = 16B row pad -> bank stride 132 dwords
#define LDS_STRIDE (E + LDS_PAD)      // 264 bf16 = 528B (16B aligned)

struct U8x32 { uint4 a, b; };         // 32-byte carrier for v16bf bit_cast

static __device__ __forceinline__ unsigned short f2bf(float f) {
    __bf16 h = (__bf16)f;
    return __builtin_bit_cast(unsigned short, h);
}
static __device__ __forceinline__ float bf2f(unsigned short s) {
    __bf16 h = __builtin_bit_cast(__bf16, s);
    return (float)h;
}

// ---------------- kernel 0: cb_sq[n] = ||codebook[n]||^2 ----------------
__global__ void vq_cbsq_kernel(const float* __restrict__ cb, float* __restrict__ cbsq) {
    int row = blockIdx.x * blockDim.x + threadIdx.x;   // 0..NE-1
    const float4* p = (const float4*)(cb + (size_t)row * E);
    float s = 0.f;
    #pragma unroll 4
    for (int i = 0; i < E / 4; ++i) {
        float4 v = p[i];
        s += v.x * v.x + v.y * v.y + v.z * v.z + v.w * v.w;
    }
    cbsq[row] = s;
}

// ---------------- kernel 1: per-segment argmin via bf16x3 WMMA ----------------
__global__ void __launch_bounds__(256, 1) vq_argmin_kernel(
    const float* __restrict__ x, const float* __restrict__ cb,
    const float* __restrict__ cbsq,
    float* __restrict__ pmin, int* __restrict__ pidx)
{
    __shared__ unsigned short lds_hi[16][LDS_STRIDE];
    __shared__ unsigned short lds_lo[16][LDS_STRIDE];

    const int tid  = threadIdx.x;
    const int lane = tid & 31;
    const int wave = tid >> 5;                 // 0..7, each wave owns a 16-row A tile
    const int seg  = blockIdx.y;
    const int m0   = blockIdx.x * 128 + wave * 16;

    // ---- A tile: 16 rows x 256, resident in registers, bf16 hi/lo split ----
    // ISA 16-bit A layout: lanes 0-15 hold M=lane; halves 0..7 -> K=kgrp+0..7,
    // halves 8..15 -> K=kgrp+16..23 ; lanes 16-31 use kgrp=8.
    const int arow = lane & 15;
    const int kgrp = (lane >> 4) * 8;
    const float* xrow = x + (size_t)(m0 + arow) * E;

    v16bf aH[8], aL[8];
    #pragma unroll
    for (int c = 0; c < 8; ++c) {
        const int base = c * 32 + kgrp;
        #pragma unroll
        for (int h = 0; h < 8; ++h) {
            float f = xrow[base + h];
            unsigned short hb = f2bf(f);
            aH[c][h] = __builtin_bit_cast(__bf16, hb);
            aL[c][h] = (__bf16)(f - bf2f(hb));
        }
        #pragma unroll
        for (int h = 0; h < 8; ++h) {
            float f = xrow[base + 16 + h];
            unsigned short hb = f2bf(f);
            aH[c][8 + h] = __builtin_bit_cast(__bf16, hb);
            aL[c][8 + h] = (__bf16)(f - bf2f(hb));
        }
    }

    // cooperative tile-fill mapping: thread -> (row, 16-col group) of codebook tile
    const int frow = tid >> 4;
    const int fcol = (tid & 15) * 16;
    const int nlane = lane & 15;

    float minval[8];
    int   minidx[8];
    #pragma unroll
    for (int r = 0; r < 8; ++r) { minval[r] = 3.4e38f; minidx[r] = 0; }

    for (int t = 0; t < TILES_PER_SEG; ++t) {
        const int n0 = seg * SEG_N + t * 16;

        // ---- convert codebook tile rows to bf16 hi/lo in LDS ----
        const float* crow = cb + (size_t)(n0 + frow) * E + fcol;
        unsigned int wh[8], wl[8];
        #pragma unroll
        for (int j = 0; j < 8; ++j) {
            float f0 = crow[2 * j], f1 = crow[2 * j + 1];
            unsigned short h0 = f2bf(f0), h1 = f2bf(f1);
            unsigned short l0 = f2bf(f0 - bf2f(h0)), l1 = f2bf(f1 - bf2f(h1));
            wh[j] = (unsigned)h0 | ((unsigned)h1 << 16);
            wl[j] = (unsigned)l0 | ((unsigned)l1 << 16);
        }
        *(uint4*)&lds_hi[frow][fcol]     = *(uint4*)&wh[0];
        *(uint4*)&lds_hi[frow][fcol + 8] = *(uint4*)&wh[4];
        *(uint4*)&lds_lo[frow][fcol]     = *(uint4*)&wl[0];
        *(uint4*)&lds_lo[frow][fcol + 8] = *(uint4*)&wl[4];

        // pull next tile toward L2/L0 while we compute (global_prefetch_b8)
        if (t + 1 < TILES_PER_SEG)
            __builtin_prefetch(crow + 16 * E, 0, 0);

        __syncthreads();

        // ---- C += A * B over K=256 : 8 chunks x 3 bf16 WMMAs (hi*hi + hi*lo + lo*hi) ----
        v8f acc = {};
        #pragma unroll
        for (int c = 0; c < 8; ++c) {
            const int koff = c * 32 + (lane >> 4) * 8;    // B layout: K=koff+0..7 and koff+16..23 per lane
            uint4 q0 = *(const uint4*)&lds_hi[nlane][koff];
            uint4 q1 = *(const uint4*)&lds_hi[nlane][koff + 16];
            U8x32 uh{q0, q1};
            v16bf bH = __builtin_bit_cast(v16bf, uh);
            uint4 p0 = *(const uint4*)&lds_lo[nlane][koff];
            uint4 p1 = *(const uint4*)&lds_lo[nlane][koff + 16];
            U8x32 ul{p0, p1};
            v16bf bL = __builtin_bit_cast(v16bf, ul);

            acc = __builtin_amdgcn_wmma_f32_16x16x32_bf16(false, aH[c], false, bH,
                                                          (short)0, acc, false, false);
            acc = __builtin_amdgcn_wmma_f32_16x16x32_bf16(false, aH[c], false, bL,
                                                          (short)0, acc, false, false);
            acc = __builtin_amdgcn_wmma_f32_16x16x32_bf16(false, aL[c], false, bH,
                                                          (short)0, acc, false, false);
        }

        // ---- running argmin of d = ||c_n||^2 - 2 x.c_n (x_sq constant per row) ----
        const float cq   = cbsq[n0 + nlane];
        const int   nidx = n0 + nlane;
        #pragma unroll
        for (int r = 0; r < 8; ++r) {
            float d = cq - 2.0f * acc[r];
            if (d < minval[r]) { minval[r] = d; minidx[r] = nidx; }
        }
        __syncthreads();
    }

    // ---- reduce across the 16-lane N group (C layout: lane holds rows m0+r+8*(lane>>4)) ----
    #pragma unroll
    for (int r = 0; r < 8; ++r) {
        float v = minval[r];
        int   i = minidx[r];
        #pragma unroll
        for (int m = 8; m >= 1; m >>= 1) {
            float ov = __shfl_xor(v, m, 32);
            int   oi = __shfl_xor(i, m, 32);
            if (ov < v || (ov == v && oi < i)) { v = ov; i = oi; }  // ties -> lowest index
        }
        if (nlane == 0) {
            const int row = m0 + r + 8 * (lane >> 4);
            pmin[seg * M_TOTAL + row] = v;
            pidx[seg * M_TOTAL + row] = i;
        }
    }
}

// ---------------- kernel 2: final argmin over segments + codebook row gather ----------------
__global__ void vq_gather_kernel(const float* __restrict__ cb,
                                 const float* __restrict__ pmin,
                                 const int* __restrict__ pidx,
                                 float* __restrict__ out)
{
    __shared__ int s_idx;
    const int m = blockIdx.x;
    if (threadIdx.x == 0) {
        float bv = pmin[m];
        int   bi = pidx[m];
        #pragma unroll
        for (int s = 1; s < NSEG; ++s) {
            float v = pmin[s * M_TOTAL + m];
            int   i = pidx[s * M_TOTAL + m];
            if (v < bv || (v == bv && i < bi)) { bv = v; bi = i; }
        }
        s_idx = bi;
    }
    __syncthreads();
    const int idx = s_idx;
    const float4* src = (const float4*)(cb + (size_t)idx * E);
    float4*       dst = (float4*)(out + (size_t)m * E);
    dst[threadIdx.x] = src[threadIdx.x];   // 64 threads x float4 = 256 floats
}

// ---------------- launcher ----------------
extern "C" void kernel_launch(void* const* d_in, const int* in_sizes, int n_in,
                              void* d_out, int out_size, void* d_ws, size_t ws_size,
                              hipStream_t stream) {
    const float* x  = (const float*)d_in[0];   // [8192, 256]
    const float* cb = (const float*)d_in[1];   // [16384, 256]
    float* out = (float*)d_out;                // [8192, 256]

    float* cbsq = (float*)d_ws;                       // NE floats
    float* pmin = cbsq + NE;                          // NSEG * M_TOTAL floats
    int*   pidx = (int*)(pmin + NSEG * M_TOTAL);      // NSEG * M_TOTAL ints

    vq_cbsq_kernel<<<NE / 256, 256, 0, stream>>>(cb, cbsq);
    vq_argmin_kernel<<<dim3(M_TOTAL / 128, NSEG), 256, 0, stream>>>(x, cb, cbsq, pmin, pidx);
    vq_gather_kernel<<<M_TOTAL, 64, 0, stream>>>(cb, pmin, pidx, out);
}
